// SelfMultiheadAttention_8796093022614
// MI455X (gfx1250) — matrix operand order repaired and checked by reading, not compiled, not run on hardware
//
#include <hip/hip_runtime.h>
#include <hip/hip_bf16.h>

typedef __bf16 bf16_t;
typedef __attribute__((ext_vector_type(16))) __bf16 bf16x16;
typedef __attribute__((ext_vector_type(8)))  __bf16 bf16x8;
typedef __attribute__((ext_vector_type(8)))  float  f32x8;
typedef __attribute__((ext_vector_type(4)))  unsigned int u32x4;
typedef __attribute__((ext_vector_type(4)))  int i32x4;
typedef __attribute__((ext_vector_type(8)))  int i32x8;

union Frag16 { bf16x16 v; bf16x8 h[2]; };

// ---------------------------------------------------------------------------
// Elementwise f32 -> bf16
// ---------------------------------------------------------------------------
__global__ void cast_f32_bf16_kernel(const float* __restrict__ src,
                                     bf16_t* __restrict__ dst, int n) {
    int i = blockIdx.x * blockDim.x + threadIdx.x;
    if (i < n) dst[i] = (bf16_t)src[i];
}

// W[K][N] (f32) -> Wt[N][K] (bf16)
__global__ void transpose_cast_kernel(const float* __restrict__ W,
                                      bf16_t* __restrict__ Wt, int K, int N) {
    int i = blockIdx.x * blockDim.x + threadIdx.x;
    if (i < K * N) {
        int k = i / N;
        int n = i - k * N;
        Wt[(size_t)n * K + k] = (bf16_t)W[(size_t)k * N + n];
    }
}

// ---------------------------------------------------------------------------
// Tensor Data Mover: 2D tile (tile_w elems wide, tile_h rows) of bf16 from
// global (row stride = stride_elems) into LDS at lds_byte_addr, with the TDM
// inserting 16 bytes of LDS padding after every 64-byte row chunk
// (pad_interval code 3 -> 16 DWORDs, pad_amount code 3 -> 4 DWORDs), so a
// 32-elem bf16 row lands in a (32+8)-elem padded LDS row.
// D# field packing per CDNA5 ISA.
// This toolchain exposes the 6-arg builtin:
//   (u32x4 g0, i32x8 g1, i32x4 g2, i32x4 g3, i32x8 extra, i32 cpol)
// ---------------------------------------------------------------------------
__device__ __forceinline__ void tdm_load_2d_bf16(const bf16_t* gptr,
                                                 unsigned lds_byte_addr,
                                                 unsigned tile_w, unsigned tile_h,
                                                 unsigned stride_elems) {
    unsigned long long ga = (unsigned long long)gptr;
    u32x4 g0;
    g0[0] = 1u;                                             // count=1, user desc
    g0[1] = lds_byte_addr;                                  // lds_addr [63:32]
    g0[2] = (unsigned)(ga & 0xFFFFFFFFu);                   // global_addr lo
    g0[3] = (unsigned)((ga >> 32) & 0x01FFFFFFu) | (2u << 30); // addr hi | type=2

    const unsigned td0 = stride_elems;       // tensor_dim0 (tiles always in-bounds)
    const unsigned td1 = 0x7FFFFFFFu;        // tensor_dim1: large
    const unsigned long long st0 = (unsigned long long)stride_elems;

    i32x8 g1;
    g1[0] = (int)((1u << 16)                 // data_size = 2 bytes
                | (1u << 20)                 // pad_enable
                | (3u << 22)                 // pad_interval: 16 DWORDs
                | (3u << 25));               // pad_amount: 4 DWORDs
    g1[1] = (int)((td0 & 0xFFFFu) << 16);                    // tensor_dim0[15:0]
    g1[2] = (int)((td0 >> 16) | ((td1 & 0xFFFFu) << 16));    // td0 hi | td1 lo
    g1[3] = (int)((td1 >> 16) | (tile_w << 16));             // td1 hi | tile_dim0
    g1[4] = (int)tile_h;                                     // tile_dim1 (dim2=0)
    g1[5] = (int)(st0 & 0xFFFFFFFFu);                        // dim0_stride lo
    g1[6] = (int)((st0 >> 32) & 0xFFFFu);                    // stride hi | dim1_stride lo
    g1[7] = 0;

    i32x4 z4 = {0, 0, 0, 0};
    i32x8 z8 = {0, 0, 0, 0, 0, 0, 0, 0};
    __builtin_amdgcn_tensor_load_to_lds(g0, g1, z4, z4, z8, 0);
}

// ---------------------------------------------------------------------------
// BF16 WMMA GEMM: C[M][N] = A[M][K] @ Bt^T + bias   (Bt stored as [N][K])
// Block tile 128(M) x 128(N), BK=32. 256 threads = 8 waves:
//   wave m-slot = wave&3 (32 rows each), wave n-slot = wave>>2 (64 cols each),
//   2x4 accumulators (16x16) per wave.
// Staging: double-buffered Tensor Data Mover DMA issued by wave 0, retired
// with s_wait_tensorcnt, compute overlapped with next tile's DMA.
// out_f32 != 0 -> write float to Cf, else bf16 to Cb.
// ---------------------------------------------------------------------------
#define GBM 128
#define GBN 128
#define GBK 32
#define GPAD 8

__global__ __launch_bounds__(256)
void gemm_bias_kernel(const bf16_t* __restrict__ A, const bf16_t* __restrict__ Bt,
                      const float* __restrict__ bias,
                      float* __restrict__ Cf, bf16_t* __restrict__ Cb,
                      int M, int N, int K, int out_f32) {
    __shared__ bf16_t As[2][GBM][GBK + GPAD];   // row stride 80 B (16B multiple)
    __shared__ bf16_t Bs[2][GBN][GBK + GPAD];

    const int t     = threadIdx.x;
    const int wave  = t >> 5;
    const int lane  = t & 31;
    const int l16   = lane & 15;
    const int lhalf = lane >> 4;
    const int wm    = wave & 3;   // 0..3, 32 rows each
    const int wn    = wave >> 2;  // 0..1, 64 cols each
    const int m0    = blockIdx.x * GBM;
    const int n0    = blockIdx.y * GBN;

    const unsigned asAddr0 = (unsigned)(unsigned long long)&As[0][0][0];
    const unsigned asAddr1 = (unsigned)(unsigned long long)&As[1][0][0];
    const unsigned bsAddr0 = (unsigned)(unsigned long long)&Bs[0][0][0];
    const unsigned bsAddr1 = (unsigned)(unsigned long long)&Bs[1][0][0];

    f32x8 acc[2][4] = {};

    const int NK = K / GBK;

    // Prologue: DMA tile 0 into buffer 0
    if (wave == 0) {
        tdm_load_2d_bf16(A  + (size_t)m0 * K, asAddr0, GBK, GBM, K);
        tdm_load_2d_bf16(Bt + (size_t)n0 * K, bsAddr0, GBK, GBN, K);
    }

    for (int i = 0; i < NK; ++i) {
        const int buf = i & 1;
        if (wave == 0) {
            if (i + 1 < NK) {
                // Issue next tile's DMA into the other buffer, then retire
                // only the current tile's two transfers (in-order per wave).
                const int k1 = (i + 1) * GBK;
                tdm_load_2d_bf16(A  + (size_t)m0 * K + k1,
                                 buf ? asAddr0 : asAddr1, GBK, GBM, K);
                tdm_load_2d_bf16(Bt + (size_t)n0 * K + k1,
                                 buf ? bsAddr0 : bsAddr1, GBK, GBN, K);
                __builtin_amdgcn_s_wait_tensorcnt((short)2);
            } else {
                __builtin_amdgcn_s_wait_tensorcnt((short)0);
            }
        }
        __syncthreads();   // release compute waves once current tile is in LDS

        // A fragments (16x32): row = lane%16; elems: K = 8h+0..7, 16+8h+0..7
        Frag16 af[2];
#pragma unroll
        for (int sub = 0; sub < 2; ++sub) {
            const int ar = wm * 32 + sub * 16 + l16;
            af[sub].h[0] = *(const bf16x8*)&As[buf][ar][8 * lhalf];
            af[sub].h[1] = *(const bf16x8*)&As[buf][ar][16 + 8 * lhalf];
        }
#pragma unroll
        for (int f = 0; f < 4; ++f) {
            // B fragment (32x16): col = lane%16; elems: K = 8h+0..7, 16+8h+0..7
            Frag16 bfr;
            const int bn = wn * 64 + f * 16 + l16;
            bfr.h[0] = *(const bf16x8*)&Bs[buf][bn][8 * lhalf];
            bfr.h[1] = *(const bf16x8*)&Bs[buf][bn][16 + 8 * lhalf];
#pragma unroll
            for (int sub = 0; sub < 2; ++sub) {
                acc[sub][f] = __builtin_amdgcn_wmma_f32_16x16x32_bf16(
                    false, af[sub].v, false, bfr.v, (short)0, acc[sub][f],
                    false, false);
            }
        }
        __syncthreads();   // all waves done reading buf before it is reused
    }

    // Epilogue: row = m0 + wm*32 + sub*16 + 8h + j, col = n0 + wn*64 + f*16 + l16
#pragma unroll
    for (int sub = 0; sub < 2; ++sub) {
        const int mbase = m0 + wm * 32 + sub * 16 + 8 * lhalf;
#pragma unroll
        for (int f = 0; f < 4; ++f) {
            const int n = n0 + wn * 64 + f * 16 + l16;
            const float bv = bias[n];
#pragma unroll
            for (int j = 0; j < 8; ++j) {
                const float v = acc[sub][f][j] + bv;
                const size_t idx = (size_t)(mbase + j) * N + n;
                if (out_f32) Cf[idx] = v;
                else         Cb[idx] = (bf16_t)v;
            }
        }
    }
}

// ---------------------------------------------------------------------------
// Flash attention over one (b, h, 64-row q-tile). 128 threads = 4 waves,
// each wave owns 16 q rows. Q/K/V/Out are [B*S, D] bf16 views, head offset
// baked into `base`. HD = 64 fixed.
// ---------------------------------------------------------------------------
__global__ __launch_bounds__(128)
void attention_kernel(const bf16_t* __restrict__ Q, const bf16_t* __restrict__ Km,
                      const bf16_t* __restrict__ V, bf16_t* __restrict__ Out,
                      int Bn, int Sn, int Hn) {
    const int HD = 64;
    const int Dn = Hn * HD;

    __shared__ bf16_t Vt[64][64 + 8];      // [hd][key], row stride 144 B
    __shared__ bf16_t Pb[4][16][64 + 8];   // per-wave P staging (C->A relayout)

    const int bh    = blockIdx.y;
    const int b     = bh / Hn;
    const int hh    = bh - b * Hn;
    const int qtile = blockIdx.x;

    const int t     = threadIdx.x;
    const int wave  = t >> 5;
    const int lane  = t & 31;
    const int l16   = lane & 15;
    const int lhalf = lane >> 4;

    const size_t base = ((size_t)b * Sn) * Dn + (size_t)hh * HD;

    // Q fragments for this wave's 16 rows, two hd-steps of 32
    Frag16 qa[2];
    {
        const int qrow = qtile * 64 + wave * 16 + l16;
        const bf16_t* qp = Q + base + (size_t)qrow * Dn;
#pragma unroll
        for (int s = 0; s < 2; ++s) {
            qa[s].h[0] = *(const bf16x8*)(qp + 32 * s + 8 * lhalf);
            qa[s].h[1] = *(const bf16x8*)(qp + 32 * s + 16 + 8 * lhalf);
        }
    }

    const float cexp = 0.125f * 1.44269504088896340736f;  // (1/sqrt(64)) * log2(e)

    float m_run[8], l_run[8];
    f32x8 oacc[4] = {};
#pragma unroll
    for (int j = 0; j < 8; ++j) { m_run[j] = -1e30f; l_run[j] = 0.0f; }

    for (int kb = 0; kb < Sn; kb += 64) {
        // ---- Stage V tile transposed: Vt[hd][key] ----
#pragma unroll
        for (int c = 0; c < 4; ++c) {
            int idx = (c * 128 + t) * 8;          // 4096 elems / (128 thr * 8)
            int key = idx >> 6;
            int hd0 = idx & 63;
            bf16x8 vv = *(const bf16x8*)(V + base + (size_t)(kb + key) * Dn + hd0);
#pragma unroll
            for (int e = 0; e < 8; ++e) Vt[hd0 + e][key] = vv[e];
        }
        __syncthreads();

        // ---- Scores: S = Q @ K^T for 64 keys (4 frags), 2 hd-steps ----
        f32x8 sa[4] = {};
#pragma unroll
        for (int kk = 0; kk < 4; ++kk) {
            const int key = kb + kk * 16 + l16;
            const bf16_t* kp = Km + base + (size_t)key * Dn;
#pragma unroll
            for (int s = 0; s < 2; ++s) {
                Frag16 kf;   // B frag: col = key, elems = hd = 32s + {8h..,16+8h..}
                kf.h[0] = *(const bf16x8*)(kp + 32 * s + 8 * lhalf);
                kf.h[1] = *(const bf16x8*)(kp + 32 * s + 16 + 8 * lhalf);
                sa[kk] = __builtin_amdgcn_wmma_f32_16x16x32_bf16(
                    false, qa[s].v, false, kf.v, (short)0, sa[kk], false, false);
            }
        }

        // ---- Online softmax (rows = elems j + 8*lhalf, cols = lanes) ----
        float tmax[8];
#pragma unroll
        for (int j = 0; j < 8; ++j) {
            float m = sa[0][j];
            m = fmaxf(m, sa[1][j]); m = fmaxf(m, sa[2][j]); m = fmaxf(m, sa[3][j]);
            tmax[j] = m;
        }
#pragma unroll
        for (int off = 1; off <= 8; off <<= 1)
#pragma unroll
            for (int j = 0; j < 8; ++j)
                tmax[j] = fmaxf(tmax[j], __shfl_xor(tmax[j], off, 32));

        float p[4][8], rowsum[8];
#pragma unroll
        for (int j = 0; j < 8; ++j) {
            const float mnew  = fmaxf(m_run[j], tmax[j]);
            const float alpha = __builtin_amdgcn_exp2f((m_run[j] - mnew) * cexp);
            m_run[j] = mnew;
            float rs = 0.0f;
#pragma unroll
            for (int kk = 0; kk < 4; ++kk) {
                p[kk][j] = __builtin_amdgcn_exp2f((sa[kk][j] - mnew) * cexp);
                rs += p[kk][j];
            }
            rowsum[j] = rs;
            l_run[j] *= alpha;
#pragma unroll
            for (int f = 0; f < 4; ++f) oacc[f][j] *= alpha;
        }
#pragma unroll
        for (int off = 1; off <= 8; off <<= 1)
#pragma unroll
            for (int j = 0; j < 8; ++j)
                rowsum[j] += __shfl_xor(rowsum[j], off, 32);
#pragma unroll
        for (int j = 0; j < 8; ++j) l_run[j] += rowsum[j];

        // ---- Write P to LDS (C layout -> memory), then PV ----
#pragma unroll
        for (int kk = 0; kk < 4; ++kk)
#pragma unroll
            for (int j = 0; j < 8; ++j)
                Pb[wave][8 * lhalf + j][kk * 16 + l16] = (bf16_t)p[kk][j];
        __syncthreads();

#pragma unroll
        for (int s2 = 0; s2 < 2; ++s2) {
            Frag16 pa;   // A frag: row = l16, elems: key = 32s2 + {8h..,16+8h..}
            pa.h[0] = *(const bf16x8*)&Pb[wave][l16][32 * s2 + 8 * lhalf];
            pa.h[1] = *(const bf16x8*)&Pb[wave][l16][32 * s2 + 16 + 8 * lhalf];
#pragma unroll
            for (int f = 0; f < 4; ++f) {
                Frag16 vb;   // B frag: col = hd, elems: key = 32s2 + {8h..,16+8h..}
                const int hd = f * 16 + l16;
                vb.h[0] = *(const bf16x8*)&Vt[hd][32 * s2 + 8 * lhalf];
                vb.h[1] = *(const bf16x8*)&Vt[hd][32 * s2 + 16 + 8 * lhalf];
                oacc[f] = __builtin_amdgcn_wmma_f32_16x16x32_bf16(
                    false, pa.v, false, vb.v, (short)0, oacc[f], false, false);
            }
        }
        __syncthreads();
    }

    // ---- Normalize and store attended (bf16) ----
#pragma unroll
    for (int f = 0; f < 4; ++f) {
        const int hd = f * 16 + l16;
#pragma unroll
        for (int j = 0; j < 8; ++j) {
            const int row = qtile * 64 + wave * 16 + 8 * lhalf + j;
            const float ov = oacc[f][j] / l_run[j];
            Out[base + (size_t)row * Dn + hd] = (bf16_t)ov;
        }
    }
}

// ---------------------------------------------------------------------------
// Host-side orchestration
// ---------------------------------------------------------------------------
extern "C" void kernel_launch(void* const* d_in, const int* in_sizes, int n_in,
                              void* d_out, int out_size, void* d_ws, size_t ws_size,
                              hipStream_t stream) {
    (void)in_sizes; (void)n_in; (void)out_size; (void)ws_size;

    const float* x  = (const float*)d_in[0];
    const float* Wq = (const float*)d_in[1];
    const float* bq = (const float*)d_in[2];
    const float* Wk = (const float*)d_in[3];
    const float* bk = (const float*)d_in[4];
    const float* Wv = (const float*)d_in[5];
    const float* bv = (const float*)d_in[6];
    const float* Wo = (const float*)d_in[7];
    const float* bo = (const float*)d_in[8];

    const int Bn = 4, Sn = 2048, Dn = 1024, Hn = 16;
    const int M = Bn * Sn;  // 8192

    size_t off = 0;
    auto alloc = [&](size_t bytes) -> void* {
        void* p = (char*)d_ws + off;
        off = (off + bytes + 255) & ~(size_t)255;
        return p;
    };
    bf16_t* Xb  = (bf16_t*)alloc((size_t)M * Dn * 2);
    bf16_t* Wqt = (bf16_t*)alloc((size_t)Dn * Dn * 2);
    bf16_t* Wkt = (bf16_t*)alloc((size_t)Dn * Dn * 2);
    bf16_t* Wvt = (bf16_t*)alloc((size_t)Dn * Dn * 2);
    bf16_t* Wot = (bf16_t*)alloc((size_t)Dn * Dn * 2);
    bf16_t* Qb  = (bf16_t*)alloc((size_t)M * Dn * 2);
    bf16_t* Kb  = (bf16_t*)alloc((size_t)M * Dn * 2);
    bf16_t* Vb  = (bf16_t*)alloc((size_t)M * Dn * 2);
    bf16_t* Att = (bf16_t*)alloc((size_t)M * Dn * 2);

    // 1) Downconvert activations and (transposed) weights to bf16
    {
        int n = M * Dn;
        cast_f32_bf16_kernel<<<(n + 255) / 256, 256, 0, stream>>>(x, Xb, n);
        int nw = Dn * Dn;
        int gw = (nw + 255) / 256;
        transpose_cast_kernel<<<gw, 256, 0, stream>>>(Wq, Wqt, Dn, Dn);
        transpose_cast_kernel<<<gw, 256, 0, stream>>>(Wk, Wkt, Dn, Dn);
        transpose_cast_kernel<<<gw, 256, 0, stream>>>(Wv, Wvt, Dn, Dn);
        transpose_cast_kernel<<<gw, 256, 0, stream>>>(Wo, Wot, Dn, Dn);
    }

    // 2) Q/K/V projections (bf16 out, f32 accumulate)
    dim3 gg(M / GBM, Dn / GBN);
    gemm_bias_kernel<<<gg, 256, 0, stream>>>(Xb, Wqt, bq, nullptr, Qb, M, Dn, Dn, 0);
    gemm_bias_kernel<<<gg, 256, 0, stream>>>(Xb, Wkt, bk, nullptr, Kb, M, Dn, Dn, 0);
    gemm_bias_kernel<<<gg, 256, 0, stream>>>(Xb, Wvt, bv, nullptr, Vb, M, Dn, Dn, 0);

    // 3) Flash attention per (b, h, q-tile)
    attention_kernel<<<dim3(Sn / 64, Bn * Hn), 128, 0, stream>>>(
        Qb, Kb, Vb, Att, Bn, Sn, Hn);

    // 4) Output projection (f32 out to d_out)
    gemm_bias_kernel<<<gg, 256, 0, stream>>>(Att, Wot, bo, (float*)d_out, nullptr,
                                             M, Dn, Dn, 1);
}
